// GlobalLocalAttention_1228360647118
// MI455X (gfx1250) — compile-verified, matrix operand order checked by reading, not run on hardware
//
#include <hip/hip_runtime.h>
#include <hip/hip_bf16.h>
#include <math.h>

// ---------------------------------------------------------------------------
// GlobalLocalAttention for MI455X (gfx1250, wave32, WMMA bf16 16x16x32,
// async global->LDS staging, ping-pong fragment pipeline).
// B=2, C=128, H=W=256, NH=16, d=8, WS=8.
// ---------------------------------------------------------------------------

typedef __attribute__((ext_vector_type(16))) __bf16 v16bf;
typedef __attribute__((ext_vector_type(8)))  float  v8f;
typedef unsigned short u16;
typedef unsigned int   u32;

#define NPIX 131072       // 2*256*256

__device__ inline u16 f2bf(float f) {
  union { float f; u32 u; } a; a.f = f;
  u32 u = a.u;
  u32 r = u + 0x7FFFu + ((u >> 16) & 1u);   // round-to-nearest-even
  return (u16)(r >> 16);
}
__device__ inline float bf2f(u16 h) {
  union { u32 u; float f; } a; a.u = ((u32)h) << 16;
  return a.f;
}

union FragAB { v16bf v; uint4 q[2]; u32 u[8]; u16 h[16]; };

// Fragment loader ("TN" layout: rows are K-contiguous; lane&15 = row,
// lane>>4 selects K half): elems 0..7 <- p[0..7], elems 8..15 <- p[16..23].
// p must be 16B aligned; emits two b128 loads.
__device__ inline void ldfrag(FragAB& f, const u16* p) {
  const uint4* q = (const uint4*)__builtin_assume_aligned((const void*)p, 16);
  f.q[0] = q[0];
  f.q[1] = q[2];
}
__device__ inline void zfrag(FragAB& f) {
  #pragma unroll
  for (int i = 0; i < 8; ++i) f.u[i] = 0u;
}
__device__ inline v8f wmma_bf(const FragAB& a, const FragAB& b, v8f c) {
  return __builtin_amdgcn_wmma_f32_16x16x32_bf16(false, a.v, false, b.v,
                                                 (short)0, c, false, false);
}

// CDNA5 async global->LDS copy (16B per lane), tracked by ASYNCcnt.
__device__ inline void async_ld16(unsigned lds_off, const void* gaddr) {
  asm volatile("global_load_async_to_lds_b128 %0, %1, off"
               :: "v"(lds_off), "v"(gaddr) : "memory");
}
__device__ inline void wait_async0() {
  asm volatile("s_wait_asynccnt 0" ::: "memory");
}

// ---------------------------------------------------------------------------
// Weight prep: fp32 -> bf16 + layout rearrangement.
// wbig: [1152][128] = rows 0-127 l2, 128-767 HGE S-stack, 768-1151 qkv.
// ---------------------------------------------------------------------------
__global__ void prep_w_k(const float* __restrict__ ca_h1, const float* __restrict__ ca_w1,
                         const float* __restrict__ ca_fh, const float* __restrict__ ca_fw,
                         const float* __restrict__ l2w,  const float* __restrict__ l1w,
                         const float* __restrict__ hgew, const float* __restrict__ fcw,
                         const float* __restrict__ qkvw, const float* __restrict__ pww,
                         u16* wca, u16* wbig, u16* wl1, u16* wfc, u16* wpw) {
  int i = blockIdx.x * 256 + threadIdx.x;
  if (i < 65536) {                                   // 4 CA matrices [o][c]
    int m = i >> 14, j = i & 16383;
    const float* s = (m == 0) ? ca_h1 : (m == 1) ? ca_w1 : (m == 2) ? ca_fh : ca_fw;
    wca[i] = f2bf(s[j]); return;
  }
  i -= 65536;
  if (i < 147456) {                                  // wbig [1152][128]
    int m = i >> 7, ii = i & 127;
    float v;
    if (m < 128) {
      v = l2w[m * 128 + ii];
    } else if (m < 768) {
      int g = (m - 128) >> 7, o = (m - 128) & 127;
      const int KY[5] = {0,0,0,1,2}, KX[5] = {0,1,2,0,2};
      v = hgew[((o * 128 + ii) * 3 + KY[g]) * 3 + KX[g]];
    } else {
      v = qkvw[(m - 768) * 128 + ii];
    }
    wbig[i] = f2bf(v); return;
  }
  i -= 147456;
  if (i < 147456) {                                  // l1: out [tap][o][i]
    int t = i >> 14, o = (i >> 7) & 127, ii = i & 127;
    int ky = t / 3, kx = t % 3;
    wl1[i] = f2bf(l1w[((o * 128 + ii) * 3 + ky) * 3 + kx]); return;
  }
  i -= 147456;
  if (i < 81920) { wfc[i] = f2bf(fcw[i]); return; }  // [o][640]
  i -= 81920;
  if (i < 16384) { wpw[i] = f2bf(pww[i]); return; }  // [o][i]
}

// Relative-position bias: bias[nh][i][j], 16*64*64
__global__ void bias_k(const float* __restrict__ rel, float* __restrict__ bias) {
  int i = blockIdx.x * 256 + threadIdx.x;
  if (i >= 16 * 4096) return;
  int nh = i >> 12, ij = i & 4095, ii = ij >> 6, jj = ij & 63;
  int dy = (ii >> 3) - (jj >> 3) + 7;
  int dx = (ii & 7)  - (jj & 7)  + 7;
  bias[i] = rel[(dy * 15 + dx) * 16 + nh];
}

// ---------------------------------------------------------------------------
// Channel attention pooling: mh[(b*256+h)][c], mw[(b*256+w)][c], bf16
// ---------------------------------------------------------------------------
__global__ void pool_k(const float* __restrict__ x, u16* __restrict__ mh,
                       u16* __restrict__ mw) {
  int i = blockIdx.x * 256 + threadIdx.x;
  if (i < 65536) {
    int c = i & 127, n = i >> 7, b = n >> 8, h = n & 255;
    const float* p = x + (((size_t)(b * 128 + c)) << 16) + (h << 8);
    float s = 0.f;
    for (int w = 0; w < 256; ++w) s += p[w];
    mh[(size_t)n * 128 + c] = f2bf(s * (1.0f / 256.0f));
  } else {
    i -= 65536;
    int c = i & 127, n = i >> 7, b = n >> 8, w = n & 255;
    const float* p = x + (((size_t)(b * 128 + c)) << 16) + w;
    float s = 0.f;
    for (int h = 0; h < 256; ++h) s += p[h << 8];
    mw[(size_t)n * 128 + c] = f2bf(s * (1.0f / 256.0f));
  }
}

// Apply CA gates, convert x (NCHW f32) -> xbf (NHWC bf16)
__global__ void scalex_k(const float* __restrict__ x, const float* __restrict__ sh,
                         const float* __restrict__ sw, u16* __restrict__ xbf) {
  size_t i = (size_t)blockIdx.x * 256 + threadIdx.x;
  int c = (int)(i & 127);
  size_t pix = i >> 7;
  int b = (int)(pix >> 16), h = (int)((pix >> 8) & 255), w = (int)(pix & 255);
  float v = x[(((size_t)(b * 128 + c)) << 16) + (pix & 65535)];
  v *= sh[((b << 8) + h) * 128 + c] * sw[((b << 8) + w) * 128 + c];
  xbf[i] = f2bf(v);
}

// ---------------------------------------------------------------------------
// Block-cooperative WMMA GEMM:  D[m][n] = sum_k A[m][k] * Bt[n][k]
// A: [M][K] bf16, Bt: [N][K] bf16.  Block = 8 waves = 128 M-rows at one
// 64-pixel N-group.  B tile (64 x K) staged once into LDS (async copies);
// fragments double-buffered (ping-pong) so WMMAs never wait on fresh loads.
// Requires M%128==0, N%64==0, K%128==0.  Dyn LDS: 64*(K+8)*2 bytes.
// emode: 0 = f32 [n][M] write (optional scale/bias/accumulate)
//        1 = bf16 [n][M] write
//        2 = sigmoid f32 [n][M] write
//        3 = f32 NCHW write (n = b*65536 + p)
//        4 = split: m<128 -> BN f32 into outf[n][128];
//                   128<=m<768 -> bf16 outb[n][640]; else bf16 outb2[n][384]
// ---------------------------------------------------------------------------
__global__ __launch_bounds__(256) void gemm_k(
    const u16* __restrict__ A, const u16* __restrict__ B,
    int M, int N, int K,
    float* __restrict__ outf, u16* __restrict__ outb, u16* __restrict__ outb2,
    const float* __restrict__ sc, const float* __restrict__ bi,
    float kscale, int emode, int acc) {
  extern __shared__ __align__(16) u16 ldsB[];   // [64][K+8]
  const int KP = K + 8;
  int tid = threadIdx.x, wave = tid >> 5, lane = tid & 31;
  int mblocks = M >> 7;
  int mb = blockIdx.x % mblocks, ng = blockIdx.x / mblocks;
  int m0 = ((mb << 3) + wave) << 4;
  int n0 = ng << 6;

  // stage B tile (async global -> LDS, 16B per lane)
  int chunks = K >> 3;                          // 16B chunks per row
  for (int idx = tid; idx < (chunks << 6); idx += 256) {
    int row = idx / chunks, c8 = (idx % chunks) << 3;
    unsigned loff = (unsigned)(size_t)&ldsB[row * KP + c8];
    async_ld16(loff, B + (size_t)(n0 + row) * K + c8);
  }
  wait_async0();
  __syncthreads();

  int r = lane & 15, hi8 = (lane >> 4) << 3;
  const u16* ap = A + (size_t)(m0 + r) * K + hi8;
  auto lb = [&](int nt, int k0) -> const u16* {
    return &ldsB[(size_t)(nt * 16 + r) * KP + k0 + hi8];
  };

  v8f accv[4] = {};
  FragAB a[4], b0[4], b1[4];
  #pragma unroll
  for (int s = 0; s < 4; ++s) ldfrag(a[s], ap + s * 32);
  #pragma unroll
  for (int nt = 0; nt < 4; ++nt) ldfrag(b0[nt], lb(nt, 0));
  #pragma unroll
  for (int nt = 0; nt < 4; ++nt) ldfrag(b1[nt], lb(nt, 32));

  for (int kc = 0; kc < K; kc += 128) {
    int kn = kc + 128;
    bool more = kn < K;
    __builtin_prefetch(ap + kn, 0, 1);
    #pragma unroll
    for (int nt = 0; nt < 4; ++nt) accv[nt] = wmma_bf(a[0], b0[nt], accv[nt]);
    #pragma unroll
    for (int nt = 0; nt < 4; ++nt) ldfrag(b0[nt], lb(nt, kc + 64));
    if (more) ldfrag(a[0], ap + kn);
    #pragma unroll
    for (int nt = 0; nt < 4; ++nt) accv[nt] = wmma_bf(a[1], b1[nt], accv[nt]);
    #pragma unroll
    for (int nt = 0; nt < 4; ++nt) ldfrag(b1[nt], lb(nt, kc + 96));
    if (more) ldfrag(a[1], ap + kn + 32);
    #pragma unroll
    for (int nt = 0; nt < 4; ++nt) accv[nt] = wmma_bf(a[2], b0[nt], accv[nt]);
    if (more) {
      #pragma unroll
      for (int nt = 0; nt < 4; ++nt) ldfrag(b0[nt], lb(nt, kn));
      ldfrag(a[2], ap + kn + 64);
    }
    #pragma unroll
    for (int nt = 0; nt < 4; ++nt) accv[nt] = wmma_bf(a[3], b1[nt], accv[nt]);
    if (more) {
      #pragma unroll
      for (int nt = 0; nt < 4; ++nt) ldfrag(b1[nt], lb(nt, kn + 32));
      ldfrag(a[3], ap + kn + 96);
    }
  }

  #pragma unroll
  for (int nt = 0; nt < 4; ++nt) {
    int n = n0 + nt * 16 + r;
    #pragma unroll
    for (int j = 0; j < 8; ++j) {
      int m = m0 + j + hi8;
      float v = accv[nt][j];
      if (emode != 4) {
        if (sc) v *= sc[m] * kscale;
        if (bi) v += bi[m];
      }
      if (emode == 0) {
        size_t o = (size_t)n * M + m;
        outf[o] = acc ? (outf[o] + v) : v;
      } else if (emode == 1) {
        outb[(size_t)n * M + m] = f2bf(v);
      } else if (emode == 2) {
        outf[(size_t)n * M + m] = 1.0f / (1.0f + __expf(-v));
      } else if (emode == 3) {
        int bb = n >> 16, p = n & 65535;
        outf[(((size_t)(bb * 128 + m)) << 16) + p] = v;
      } else {                                  // emode 4: fused l2/hge/qkv
        if (m < 128)
          outf[((size_t)n << 7) + m] = v * (sc[m] * kscale) + bi[m];
        else if (m < 768)
          outb[(size_t)n * 640 + (m - 128)] = f2bf(v);
        else
          outb2[(size_t)n * 384 + (m - 768)] = f2bf(v);
      }
    }
  }
}

// ---------------------------------------------------------------------------
// Implicit-GEMM 3x3 conv (zero pad 1): local += bn(conv3x3(xbf, l1))
// A: [9][128][128] bf16, X: NHWC bf16.
// Block = 8 waves = full M=128 at one 64-pixel N-group (one image row).
// Halo tile (3 rows x 66 px x 128 ch, OOB zeroed) staged via async copies.
// Dyn LDS: 3*66*136*2 = 53856 bytes.
// ---------------------------------------------------------------------------
__global__ __launch_bounds__(256) void gemm_conv3_k(
    const u16* __restrict__ A, const u16* __restrict__ X,
    float* __restrict__ local, const float* __restrict__ g,
    const float* __restrict__ bvec, float kscale) {
  extern __shared__ __align__(16) u16 ldsX[];   // [(ry*66+cx)*136 + ch]
  const int CP = 136;
  int tid = threadIdx.x, wave = tid >> 5, lane = tid & 31;
  int n0 = blockIdx.x << 6;
  int b = n0 >> 16, h = (n0 >> 8) & 255, w0 = n0 & 255;

  // stage halo: rows h-1..h+1, cols w0-1..w0+64, 128 ch (16 x 16B chunks)
  for (int idx = tid; idx < 3 * 66 * 16; idx += 256) {
    int ry = idx / (66 * 16), rem = idx % (66 * 16);
    int cx = rem >> 4, c8 = (rem & 15) << 3;
    int hh = h - 1 + ry, w = w0 - 1 + cx;
    unsigned loff = (unsigned)(size_t)&ldsX[(ry * 66 + cx) * CP + c8];
    if ((unsigned)hh < 256u && (unsigned)w < 256u) {
      async_ld16(loff, X + (((size_t)((b * 256 + hh) * 256 + w)) << 7) + c8);
    } else {
      *(uint4*)&ldsX[(ry * 66 + cx) * CP + c8] = uint4{0u, 0u, 0u, 0u};
    }
  }
  wait_async0();
  __syncthreads();

  int r = lane & 15, hi8 = (lane >> 4) << 3;
  int m0 = wave << 4;
  auto lbx = [&](int dy, int dx, int nt, int k0) -> const u16* {
    return &ldsX[(size_t)(dy * 66 + nt * 16 + r + dx) * CP + k0 + hi8];
  };
  auto lA = [&](FragAB& f, int t, int s) {
    ldfrag(f, A + (size_t)((t * 128 + m0 + r) * 128) + s * 32 + hi8);
  };

  v8f accv[4] = {};
  FragAB a[4], b0[4], b1[4];
  #pragma unroll
  for (int s = 0; s < 4; ++s) lA(a[s], 0, s);
  #pragma unroll
  for (int nt = 0; nt < 4; ++nt) ldfrag(b0[nt], lbx(0, 0, nt, 0));
  #pragma unroll
  for (int nt = 0; nt < 4; ++nt) ldfrag(b1[nt], lbx(0, 0, nt, 32));

  for (int t = 0; t < 9; ++t) {
    int dy = t / 3, dx = t % 3;
    int tn = t + 1;
    bool more = tn < 9;
    int dyn = tn / 3, dxn = tn % 3;
    #pragma unroll
    for (int nt = 0; nt < 4; ++nt) accv[nt] = wmma_bf(a[0], b0[nt], accv[nt]);
    #pragma unroll
    for (int nt = 0; nt < 4; ++nt) ldfrag(b0[nt], lbx(dy, dx, nt, 64));
    if (more) lA(a[0], tn, 0);
    #pragma unroll
    for (int nt = 0; nt < 4; ++nt) accv[nt] = wmma_bf(a[1], b1[nt], accv[nt]);
    #pragma unroll
    for (int nt = 0; nt < 4; ++nt) ldfrag(b1[nt], lbx(dy, dx, nt, 96));
    if (more) lA(a[1], tn, 1);
    #pragma unroll
    for (int nt = 0; nt < 4; ++nt) accv[nt] = wmma_bf(a[2], b0[nt], accv[nt]);
    if (more) {
      #pragma unroll
      for (int nt = 0; nt < 4; ++nt) ldfrag(b0[nt], lbx(dyn, dxn, nt, 0));
      lA(a[2], tn, 2);
    }
    #pragma unroll
    for (int nt = 0; nt < 4; ++nt) accv[nt] = wmma_bf(a[3], b1[nt], accv[nt]);
    if (more) {
      #pragma unroll
      for (int nt = 0; nt < 4; ++nt) ldfrag(b1[nt], lbx(dyn, dxn, nt, 32));
      lA(a[3], tn, 3);
    }
  }

  #pragma unroll
  for (int nt = 0; nt < 4; ++nt) {
    int n = n0 + nt * 16 + r;
    #pragma unroll
    for (int j = 0; j < 8; ++j) {
      int m = m0 + j + hi8;
      float v = accv[nt][j] * (g[m] * kscale) + bvec[m];
      local[((size_t)n << 7) + m] += v;
    }
  }
}

// ---------------------------------------------------------------------------
// Depthwise HGE stencils on the 5 channel-mixed groups (zero pad).
// y, feats: [pixel][640] bf16
// ---------------------------------------------------------------------------
__global__ void hge_dw_k(const u16* __restrict__ y, u16* __restrict__ feats) {
  size_t i = (size_t)blockIdx.x * 256 + threadIdx.x;
  int c = (int)(i & 127);
  size_t pix = i >> 7;
  int b = (int)(pix >> 16), h = (int)((pix >> 8) & 255), w = (int)(pix & 255);
  auto rd = [&](int g, int dh, int dw_) -> float {
    int hh = h + dh, ww = w + dw_;
    if ((unsigned)hh >= 256u || (unsigned)ww >= 256u) return 0.f;
    return bf2f(y[((size_t)((b * 256 + hh) * 256 + ww)) * 640 + g * 128 + c]);
  };
  float f0 = rd(0,-1,-1) + 2.f*rd(0,-1,0) + rd(0,-1,1)
           - rd(0, 1,-1) - 2.f*rd(0, 1,0) - rd(0, 1,1);
  float f1 = 2.f*rd(1,-1,-1) + rd(1,-1,0) + rd(1,0,-1)
           - rd(1,0,1) - rd(1,1,0) - 2.f*rd(1,1,1);
  float f2 = rd(2,-1,-1) - rd(2,-1,1) + 2.f*rd(2,0,-1) - 2.f*rd(2,0,1)
           + rd(2,1,-1) - rd(2,1,1);
  float f3 = -rd(3,-1,0) - 2.f*rd(3,-1,1) + rd(3,0,-1) - rd(3,0,1)
           + 2.f*rd(3,1,-1) + rd(3,1,0);
  float f4 = 4.f*rd(4,0,0)
           - 0.5f *(rd(4,-1,0)+rd(4,0,-1)+rd(4,0,1)+rd(4,1,0))
           - 0.25f*(rd(4,-2,0)+rd(4,-1,-1)+rd(4,-1,1)+rd(4,0,-2)
                   +rd(4,0,2)+rd(4,1,-1)+rd(4,1,1)+rd(4,2,0));
  size_t o = pix * 640 + c;
  feats[o]       = f2bf(f0);
  feats[o + 128] = f2bf(f1);
  feats[o + 256] = f2bf(f2);
  feats[o + 384] = f2bf(f3);
  feats[o + 512] = f2bf(f4);
}

// ---------------------------------------------------------------------------
// Window attention: one wave per (window, head). 8 waves/block.
// qkv: [pixel][384] bf16 (q:0-127, k:128-255, v:256-383, ch = nh*8+d)
// LDS per wave: 64x64 f32 scores + 64x64 bf16 probs. (dyn LDS 196608 B)
// ---------------------------------------------------------------------------
__global__ __launch_bounds__(256) void attn_k(const u16* __restrict__ qkv,
                                              const float* __restrict__ bias,
                                              float* __restrict__ obuf) {
  extern __shared__ __align__(16) char smem[];
  int wave = threadIdx.x >> 5, lane = threadIdx.x & 31;
  int r = lane & 15, hi = lane >> 4;
  float* Sw = (float*)smem + (size_t)wave * 4096;
  u16*   Pw = (u16*)(smem + 8 * 4096 * 4) + (size_t)wave * 4096;
  int head = blockIdx.y * 8 + wave;
  int widx = blockIdx.x;
  int b = widx >> 10, rem = widx & 1023, hh = rem >> 5, ww = rem & 31;
  int pixbase = (b * 256 + hh * 8) * 256 + ww * 8;
  auto px = [&](int t) { return pixbase + (t >> 3) * 256 + (t & 7); };

  // --- S = Q K^T (d=8 zero-padded into K=32) ---
  FragAB kf[4];
  #pragma unroll
  for (int ni = 0; ni < 4; ++ni) {
    zfrag(kf[ni]);
    if (hi == 0) {
      const uint4* p = (const uint4*)__builtin_assume_aligned(
          (const void*)(qkv + (size_t)px(ni * 16 + r) * 384 + 128 + head * 8), 16);
      kf[ni].q[0] = p[0];
    }
  }
  for (int mi = 0; mi < 4; ++mi) {
    FragAB qf; zfrag(qf);
    if (hi == 0) {
      const uint4* p = (const uint4*)__builtin_assume_aligned(
          (const void*)(qkv + (size_t)px(mi * 16 + r) * 384 + head * 8), 16);
      qf.q[0] = p[0];
    }
    #pragma unroll
    for (int ni = 0; ni < 4; ++ni) {
      v8f acc = {};
      acc = wmma_bf(qf, kf[ni], acc);
      #pragma unroll
      for (int j = 0; j < 8; ++j)
        Sw[(mi * 16 + j + hi * 8) * 64 + ni * 16 + r] = acc[j];
    }
  }
  __syncthreads();

  // --- softmax rows (2 rows per lane) ---
  const float scl = 0.35355339059327373f;  // 8^-0.5
  const float* brow = bias + (size_t)head * 4096;
  for (int row = lane; row < 64; row += 32) {
    float mx = -1e30f;
    for (int j = 0; j < 64; ++j) {
      float v = Sw[row * 64 + j] * scl + brow[row * 64 + j];
      Sw[row * 64 + j] = v;
      mx = fmaxf(mx, v);
    }
    float s = 0.f;
    for (int j = 0; j < 64; ++j) {
      float e = __expf(Sw[row * 64 + j] - mx);
      s += e;
      Sw[row * 64 + j] = e;
    }
    float inv = 1.0f / s;
    for (int j = 0; j < 64; ++j)
      Pw[row * 64 + j] = f2bf(Sw[row * 64 + j] * inv);
  }
  __syncthreads();

  // --- O = P V  (M=64, K=64, N=8 padded to 16) ---
  FragAB vf[2];
  #pragma unroll
  for (int s = 0; s < 2; ++s) {
    int kk = s * 32;
    #pragma unroll
    for (int e = 0; e < 8; ++e) {
      int k1 = kk + hi * 8 + e;
      int k2 = kk + 16 + hi * 8 + e;
      u16 a = 0, b2 = 0;
      if (r < 8) {
        a  = qkv[(size_t)px(k1) * 384 + 256 + head * 8 + r];
        b2 = qkv[(size_t)px(k2) * 384 + 256 + head * 8 + r];
      }
      vf[s].h[e] = a; vf[s].h[8 + e] = b2;
    }
  }
  for (int mi = 0; mi < 4; ++mi) {
    v8f acc = {};
    #pragma unroll
    for (int s = 0; s < 2; ++s) {
      FragAB pf;
      ldfrag(pf, &Pw[(mi * 16 + r) * 64 + s * 32 + hi * 8]);
      acc = wmma_bf(pf, vf[s], acc);
    }
    if (r < 8) {
      #pragma unroll
      for (int j = 0; j < 8; ++j) {
        int tok = mi * 16 + j + hi * 8;
        obuf[((size_t)px(tok) << 7) + head * 8 + r] = acc[j];
      }
    }
  }
}

// ---------------------------------------------------------------------------
// out_pre = avgpool_x(o) + avgpool_y(o) + local   (reflect (0,1) + zero pad 3)
// ---------------------------------------------------------------------------
__global__ void combine_k(const float* __restrict__ o, const float* __restrict__ local,
                          float* __restrict__ outpre) {
  size_t i = (size_t)blockIdx.x * 256 + threadIdx.x;
  int c = (int)(i & 127);
  size_t pix = i >> 7;
  int b = (int)(pix >> 16), h = (int)((pix >> 8) & 255), w = (int)(pix & 255);
  float sx = 0.f, sy = 0.f;
  #pragma unroll
  for (int t = 0; t < 8; ++t) {
    int rr = h - 3 + t;
    if (rr >= 0 && rr <= 256) {
      int r2 = (rr == 256) ? 254 : rr;
      sx += o[(((size_t)((b * 256 + r2) * 256 + w)) << 7) + c];
    }
    int cc = w - 3 + t;
    if (cc >= 0 && cc <= 256) {
      int c2 = (cc == 256) ? 254 : cc;
      sy += o[(((size_t)((b * 256 + h) * 256 + c2)) << 7) + c];
    }
  }
  outpre[i] = 0.125f * (sx + sy) + local[i];
}

// ---------------------------------------------------------------------------
// Reflect-pad (0,1)x(0,1) + 8x8 depthwise conv (pad 3) + BN -> bf16
// ---------------------------------------------------------------------------
__global__ void dwconv_k(const float* __restrict__ inp, const float* __restrict__ wdw,
                         const float* __restrict__ g, const float* __restrict__ bb,
                         u16* __restrict__ outb, float kscale) {
  size_t i = (size_t)blockIdx.x * 256 + threadIdx.x;
  int c = (int)(i & 127);
  size_t pix = i >> 7;
  int b = (int)(pix >> 16), h = (int)((pix >> 8) & 255), w = (int)(pix & 255);
  float s = 0.f;
  for (int ky = 0; ky < 8; ++ky) {
    int rr = h - 3 + ky;
    if (rr < 0 || rr > 256) continue;
    int r2 = (rr == 256) ? 254 : rr;
    for (int kx = 0; kx < 8; ++kx) {
      int cc = w - 3 + kx;
      if (cc < 0 || cc > 256) continue;
      int c2 = (cc == 256) ? 254 : cc;
      s += inp[(((size_t)((b * 256 + r2) * 256 + c2)) << 7) + c] * wdw[c * 64 + ky * 8 + kx];
    }
  }
  float v = s * (g[c] * kscale) + bb[c];
  outb[i] = f2bf(v);
}

// ---------------------------------------------------------------------------
// Host-side orchestration
// ---------------------------------------------------------------------------
extern "C" void kernel_launch(void* const* d_in, const int* in_sizes, int n_in,
                              void* d_out, int out_size, void* d_ws, size_t ws_size,
                              hipStream_t stream) {
  (void)in_sizes; (void)n_in; (void)out_size; (void)ws_size;
  const float* x      = (const float*)d_in[0];
  const float* ca_h1  = (const float*)d_in[1];
  const float* ca_w1  = (const float*)d_in[2];
  const float* ca_fh  = (const float*)d_in[3];
  const float* ca_fw  = (const float*)d_in[4];
  const float* l1_w   = (const float*)d_in[5];
  const float* l1_g   = (const float*)d_in[6];
  const float* l1_b   = (const float*)d_in[7];
  const float* l2_w   = (const float*)d_in[8];
  const float* l2_g   = (const float*)d_in[9];
  const float* l2_b   = (const float*)d_in[10];
  const float* hge_w  = (const float*)d_in[11];
  const float* fc_w   = (const float*)d_in[12];
  const float* fc_b   = (const float*)d_in[13];
  const float* qkv_w  = (const float*)d_in[14];
  const float* rel_t  = (const float*)d_in[15];
  const float* pdw    = (const float*)d_in[16];
  const float* pg     = (const float*)d_in[17];
  const float* pb     = (const float*)d_in[18];
  const float* ppw    = (const float*)d_in[19];
  float* out          = (float*)d_out;

  char* ws = (char*)d_ws;
  u16*   wca    = (u16*)(ws + 0);           // 4x[128][128]
  u16*   wbig   = (u16*)(ws + 131072);      // [1152][128]: l2 | hge | qkv
  u16*   wl1    = (u16*)(ws + 425984);      // [9][128][128]
  u16*   wfc    = (u16*)(ws + 720896);      // [128][640]
  u16*   wpw    = (u16*)(ws + 884736);      // [128][128]
  float* biasb  = (float*)(ws + 917504);    // [16][64][64]
  u16*   mh     = (u16*)(ws + 1179648);     // [512][128] bf16
  u16*   mw     = (u16*)(ws + 1310720);
  u16*   t1     = (u16*)(ws + 1441792);     // [512][128] bf16 (reused h then w)
  float* sh     = (float*)(ws + 1572864);   // [512][128] f32
  float* sw     = (float*)(ws + 1835008);
  u16*   xbf    = (u16*)(ws + 2097152);             // [NPIX][128] bf16
  float* local  = (float*)(ws + 35651584);          // [NPIX][128] f32
  char*  regA   = ws + 102760448;                   // 160 MiB
  char*  regB   = ws + 270532608;                   // 160 MiB
  u16*   qkvb   = (u16*)(ws + 438304768);           // [NPIX][384] bf16
  u16*   yhge   = (u16*)regA;                       // [NPIX][640] bf16
  u16*   dwout  = (u16*)regA;                       // [NPIX][128] bf16 (after hge_fc)
  u16*   feats  = (u16*)regB;                       // [NPIX][640] bf16
  float* obuf   = (float*)regB;                     // [NPIX][128] f32 (after hge_fc)
  float* outpre = (float*)(regB + 67108864);        // [NPIX][128] f32

  const float kscale = 1.0f / sqrtf(1.0f + 1e-5f);  // BN 1/sqrt(1+eps)
  const unsigned LDS128 = 64u * (128 + 8) * 2;      // 17408 B (K=128)
  const unsigned LDS640 = 64u * (640 + 8) * 2;      // 82944 B (K=640)
  const unsigned LDSC3  = 3u * 66 * 136 * 2;        // 53856 B

  // 1) weight prep + bias table
  prep_w_k<<<1792, 256, 0, stream>>>(ca_h1, ca_w1, ca_fh, ca_fw, l2_w, l1_w,
                                     hge_w, fc_w, qkv_w, ppw,
                                     wca, wbig, wl1, wfc, wpw);
  bias_k<<<256, 256, 0, stream>>>(rel_t, biasb);

  // 2) channel attention
  pool_k<<<512, 256, 0, stream>>>(x, mh, mw);
  gemm_k<<<8, 256, LDS128, stream>>>(wca,         mh, 128, 512, 128,
                                     nullptr, t1, nullptr, nullptr, nullptr, 1.f, 1, 0);
  gemm_k<<<8, 256, LDS128, stream>>>(wca + 32768, t1, 128, 512, 128,
                                     sh, nullptr, nullptr, nullptr, nullptr, 1.f, 2, 0);
  gemm_k<<<8, 256, LDS128, stream>>>(wca + 16384, mw, 128, 512, 128,
                                     nullptr, t1, nullptr, nullptr, nullptr, 1.f, 1, 0);
  gemm_k<<<8, 256, LDS128, stream>>>(wca + 49152, t1, 128, 512, 128,
                                     sw, nullptr, nullptr, nullptr, nullptr, 1.f, 2, 0);
  scalex_k<<<65536, 256, 0, stream>>>(x, sh, sw, xbf);

  // 3) fused l2/HGE-mix/qkv GEMM (M=1152, one pass over xbf)
  gemm_k<<<2048 * 9, 256, LDS128, stream>>>(wbig, xbf, 1152, NPIX, 128,
                                            local, yhge, qkvb, l2_g, l2_b,
                                            kscale, 4, 0);
  // 3b) l1 3x3 implicit GEMM (accumulate into local)
  gemm_conv3_k<<<2048, 256, LDSC3, stream>>>(wl1, xbf, local, l1_g, l1_b, kscale);

  // 4) HGE depthwise stencils -> hge_fc (K=640, accumulate into local)
  hge_dw_k<<<65536, 256, 0, stream>>>(yhge, feats);
  gemm_k<<<2048, 256, LDS640, stream>>>(wfc, feats, 128, NPIX, 640,
                                        local, nullptr, nullptr, nullptr, fc_b,
                                        1.f, 0, 1);

  // 5) attention
  attn_k<<<dim3(2048, 2), 256, 196608, stream>>>(qkvb, biasb, obuf);

  // 6) combine + projection
  combine_k<<<65536, 256, 0, stream>>>(obuf, local, outpre);
  dwconv_k<<<65536, 256, 0, stream>>>(outpre, pdw, pg, pb, dwout, kscale);
  gemm_k<<<2048, 256, LDS128, stream>>>(wpw, dwout, 128, NPIX, 128,
                                        out, nullptr, nullptr, nullptr, nullptr,
                                        1.f, 3, 0);
}
